// SICModel2_32667521253810
// MI455X (gfx1250) — matrix operand order repaired and checked by reading, not compiled, not run on hardware
//
#include <hip/hip_runtime.h>
#include <stdint.h>

typedef __attribute__((ext_vector_type(16))) __bf16 v16bf;
typedef __attribute__((ext_vector_type(8)))  float  v8f;
typedef __attribute__((ext_vector_type(4)))  unsigned int v4u;
typedef __attribute__((ext_vector_type(8)))  int v8i;
typedef __attribute__((ext_vector_type(4)))  int v4i;

#define HDIM 1024
#define SLEN 1024
#define BATCH 16
#define KIDX 256
#define MROWS (BATCH * KIDX)   /* 4096 gathered rows per (side, start/end) */
#define KCHUNK 64              /* K staged per LDS chunk */
#define NCHUNK (HDIM / KCHUNK) /* 16 */

#if __has_builtin(__builtin_amdgcn_tensor_load_to_lds) && __has_builtin(__builtin_amdgcn_s_wait_tensorcnt)
#define HAVE_TDM 1
#else
#define HAVE_TDM 0
#endif

__device__ __forceinline__ unsigned short f32_to_bf16_rne(float f) {
  union { float f; uint32_t u; } c; c.f = f;
  uint32_t u = c.u;
  u += 0x7FFFu + ((u >> 16) & 1u);   // round-to-nearest-even
  return (unsigned short)(u >> 16);
}

// ---------------------------------------------------------------------------
// Kernel 1: gather rows of hidden_states at start/end indices -> bf16 A bufs
// ---------------------------------------------------------------------------
__global__ void gather_rows_kernel(const float* __restrict__ x,
                                   const int* __restrict__ prem_idx,
                                   const int* __restrict__ hypo_idx,
                                   unsigned short* __restrict__ ws_a) {
  const int row  = blockIdx.x;   // 0..4095  (b*256 + k)
  const int se   = blockIdx.y;   // 0 = start, 1 = end
  const int side = blockIdx.z;   // 0 = prem, 1 = hypo
  const int* idx = side ? hypo_idx : prem_idx;
  const int b = row >> 8;
  const int k = row & 255;
  const int s = idx[(b * 2 + se) * KIDX + k];   // idx layout (B,1,2,K)

  const float* src = x + ((size_t)b * SLEN + (size_t)s) * HDIM;
  unsigned short* dst = ws_a + ((size_t)(side * 2 + se)) * (size_t)MROWS * HDIM
                             + (size_t)row * HDIM;
  const int t = threadIdx.x * 4;
  float4 v = *(const float4*)(src + t);
  ushort4 o;
  o.x = f32_to_bf16_rne(v.x);
  o.y = f32_to_bf16_rne(v.y);
  o.z = f32_to_bf16_rne(v.z);
  o.w = f32_to_bf16_rne(v.w);
  *(ushort4*)(dst + t) = o;
}

// ---------------------------------------------------------------------------
// Kernel 2: convert 8 weight matrices (H x H, f32) to bf16
// ---------------------------------------------------------------------------
struct WPtrs { const float* w[8]; };

__global__ void convert_w_kernel(WPtrs p, unsigned short* __restrict__ ws_w) {
  const int mat = blockIdx.y;
  const float* src = p.w[mat];
  unsigned short* dst = ws_w + (size_t)mat * (size_t)HDIM * HDIM;
  const int t = (blockIdx.x * 256 + threadIdx.x) * 4;
  float4 v = *(const float4*)(src + t);
  ushort4 o;
  o.x = f32_to_bf16_rne(v.x);
  o.y = f32_to_bf16_rne(v.y);
  o.z = f32_to_bf16_rne(v.z);
  o.w = f32_to_bf16_rne(v.w);
  *(ushort4*)(dst + t) = o;
}

// ---------------------------------------------------------------------------
// TDM helper: 2D tile load (rows x cols bf16, row stride HDIM) global -> LDS.
// D# bitfields per CDNA5 ISA ch.8 (group0: count/lds/global/type,
// group1: data_size, tensor dims, tile dims, dim0 stride).
// ---------------------------------------------------------------------------
#if HAVE_TDM
__device__ __forceinline__ void tdm_load_2d(unsigned lds_off,
                                            const unsigned short* gptr,
                                            int rows, int cols) {
  unsigned long long ga = (unsigned long long)(uintptr_t)gptr;
  v4u g0;
  g0.x = 1u;                                    // count=1 (valid), user mode
  g0.y = lds_off;                               // lds_addr (bytes)
  g0.z = (unsigned)(ga & 0xFFFFFFFFu);          // global_addr[31:0]
  g0.w = (unsigned)((ga >> 32) & 0x01FFFFFFu)   // global_addr[56:32]
       | (2u << 30);                            // type = 2 ("image")
  v8i g1;
  g1[0] = (1 << 16);                            // data_size=1 -> 2 bytes
  g1[1] = (int)((HDIM & 0xFFFF) << 16);         // tensor_dim0[15:0]
  g1[2] = (int)(((HDIM >> 16) & 0xFFFF) | ((MROWS & 0xFFFF) << 16)); // dim0 hi | dim1 lo
  g1[3] = (int)((((unsigned)MROWS >> 16) & 0xFFFF) | (((unsigned)cols & 0xFFFF) << 16)); // dim1 hi | tile_dim0
  g1[4] = rows & 0xFFFF;                        // tile_dim1 (tile_dim2 = 0)
  g1[5] = HDIM;                                 // tensor_dim0_stride[31:0]
  g1[6] = 0;
  g1[7] = 0;
  v4i z4 = {0, 0, 0, 0};
#if __clang_major__ >= 23
  v8i z8 = {0, 0, 0, 0, 0, 0, 0, 0};
  __builtin_amdgcn_tensor_load_to_lds(g0, g1, z4, z4, z8, 0);
#else
  __builtin_amdgcn_tensor_load_to_lds(g0, g1, z4, z4, 0);
#endif
}
#endif

// ---------------------------------------------------------------------------
// Kernel 3: span GEMM. Block = 8 waves, all sharing one 32-row m-pair.
// Each wave: 32m x 16n tile = 2 m-subtiles x 6 accumulators = 12 WMMA / K-step.
// A (start+end rows) staged in LDS via TDM double-buffering; B from global.
// ---------------------------------------------------------------------------
union FragBf { v16bf v; uint4 q[2]; };

#define WMMA_BF16(A, B, C) \
  __builtin_amdgcn_wmma_f32_16x16x32_bf16(false, (A), false, (B), (short)0, (C), false, false)

__global__ __launch_bounds__(256)
void span_gemm_kernel(const unsigned short* __restrict__ ws_a,
                      const unsigned short* __restrict__ ws_w,
                      const float* __restrict__ bp1, const float* __restrict__ bp2,
                      const float* __restrict__ bp4,
                      const float* __restrict__ bh1, const float* __restrict__ bh2,
                      const float* __restrict__ bh4,
                      float* __restrict__ out) {
  // [buf][src(start/end)][row 0..31][k 0..63]
  __shared__ __align__(16) unsigned short sA[2][2][32][KCHUNK];

  const int lane  = threadIdx.x & 31;
  const int wave  = threadIdx.x >> 5;
  const int side  = blockIdx.z;
  const int mpair = blockIdx.y;             // 0..127 : 32 rows each
  const int ntile = blockIdx.x * 8 + wave;  // 0..63
  const int m0 = mpair * 32;
  const int n0 = ntile * 16;

  const unsigned short* As = ws_a + (size_t)(side * 2 + 0) * (size_t)MROWS * HDIM;
  const unsigned short* Ae = ws_a + (size_t)(side * 2 + 1) * (size_t)MROWS * HDIM;
  const unsigned short* Wb = ws_w + (size_t)side * 4u * (size_t)HDIM * HDIM;
  const unsigned short* W1 = Wb;
  const unsigned short* W2 = Wb + (size_t)1 * HDIM * HDIM;
  const unsigned short* W3 = Wb + (size_t)2 * HDIM * HDIM;
  const unsigned short* W4 = Wb + (size_t)3 * HDIM * HDIM;

  const int half = lane >> 4;   // K sub-block selector (per ISA A/B layouts)
  const int l15  = lane & 15;   // A row within tile / B column within tile
  const size_t brow = (size_t)(n0 + l15) * HDIM;

  // acc[msub][0..5] = c1, c2, c3s, c3e, c4s, c4e
  v8f acc[2][6];
#pragma unroll
  for (int t = 0; t < 2; ++t)
#pragma unroll
    for (int j = 0; j < 6; ++j) acc[t][j] = (v8f){};

#if HAVE_TDM
  // ---- prologue: stage chunk 0 into buf 0 ----
  if (wave == 0) {
    tdm_load_2d((unsigned)(uintptr_t)&sA[0][0][0][0], As + (size_t)m0 * HDIM, 32, KCHUNK);
    tdm_load_2d((unsigned)(uintptr_t)&sA[0][1][0][0], Ae + (size_t)m0 * HDIM, 32, KCHUNK);
  }
#endif

  for (int ic = 0; ic < NCHUNK; ++ic) {
    const int kc = ic * KCHUNK;
#if HAVE_TDM
    const int cur = ic & 1;
    if (wave == 0) {
      if (ic + 1 < NCHUNK) {
        const int kn = kc + KCHUNK;
        tdm_load_2d((unsigned)(uintptr_t)&sA[cur ^ 1][0][0][0],
                    As + (size_t)m0 * HDIM + kn, 32, KCHUNK);
        tdm_load_2d((unsigned)(uintptr_t)&sA[cur ^ 1][1][0][0],
                    Ae + (size_t)m0 * HDIM + kn, 32, KCHUNK);
        __builtin_amdgcn_s_wait_tensorcnt(2);   // chunk ic complete (in-order)
      } else {
        __builtin_amdgcn_s_wait_tensorcnt(0);
      }
    }
    __syncthreads();   // chunk ic visible to all waves
#else
    const int cur = 0;
    // cooperative fallback: 256 threads copy 2 x 32 x 64 bf16 (8 elems each x2)
#pragma unroll
    for (int j = 0; j < 2; ++j) {
      const int e = (threadIdx.x + j * 256) * 8;   // element 0..4095
      const int src_sel = e >> 11;                 // 0=start, 1=end
      const int rem = e & 2047;
      const int row = rem >> 6;
      const int k = rem & 63;
      const unsigned short* g = (src_sel ? Ae : As)
                              + (size_t)(m0 + row) * HDIM + kc + k;
      *(uint4*)&sA[cur][src_sel][row][k] = *(const uint4*)g;
    }
    __syncthreads();
#endif

#pragma unroll
    for (int ks = 0; ks < 2; ++ks) {
      const int kl = ks * 32;          // chunk-local K base
      const int ka = kl + half * 8;    // A frag: halves take K {+0..7,+16..23}/{+8..15,+24..31}
      // A fragments from LDS: [0]=As rows 0-15, [1]=As rows 16-31,
      //                       [2]=Ae rows 0-15, [3]=Ae rows 16-31
      FragBf fa0, fa1, fa2, fa3;
      fa0.q[0] = *(const uint4*)&sA[cur][0][l15][ka];
      fa0.q[1] = *(const uint4*)&sA[cur][0][l15][ka + 16];
      fa1.q[0] = *(const uint4*)&sA[cur][0][16 + l15][ka];
      fa1.q[1] = *(const uint4*)&sA[cur][0][16 + l15][ka + 16];
      fa2.q[0] = *(const uint4*)&sA[cur][1][l15][ka];
      fa2.q[1] = *(const uint4*)&sA[cur][1][l15][ka + 16];
      fa3.q[0] = *(const uint4*)&sA[cur][1][16 + l15][ka];
      fa3.q[1] = *(const uint4*)&sA[cur][1][16 + l15][ka + 16];

      // B fragments from global: halves take K {+0..7,+16..23}/{+8..15,+24..31}
      const int kb = kc + kl + half * 8;
      FragBf fb1, fb2, fb3, fb4;
      fb1.q[0] = *(const uint4*)(W1 + brow + kb);
      fb1.q[1] = *(const uint4*)(W1 + brow + kb + 16);
      fb2.q[0] = *(const uint4*)(W2 + brow + kb);
      fb2.q[1] = *(const uint4*)(W2 + brow + kb + 16);
      fb3.q[0] = *(const uint4*)(W3 + brow + kb);
      fb3.q[1] = *(const uint4*)(W3 + brow + kb + 16);
      fb4.q[0] = *(const uint4*)(W4 + brow + kb);
      fb4.q[1] = *(const uint4*)(W4 + brow + kb + 16);

      // m-subtile 0 (rows m0..m0+15)
      acc[0][0] = WMMA_BF16(fa0.v, fb1.v, acc[0][0]);
      acc[0][1] = WMMA_BF16(fa2.v, fb2.v, acc[0][1]);
      acc[0][2] = WMMA_BF16(fa0.v, fb3.v, acc[0][2]);
      acc[0][3] = WMMA_BF16(fa2.v, fb3.v, acc[0][3]);
      acc[0][4] = WMMA_BF16(fa0.v, fb4.v, acc[0][4]);
      acc[0][5] = WMMA_BF16(fa2.v, fb4.v, acc[0][5]);
      // m-subtile 1 (rows m0+16..m0+31) — reuses all four B fragments
      acc[1][0] = WMMA_BF16(fa1.v, fb1.v, acc[1][0]);
      acc[1][1] = WMMA_BF16(fa3.v, fb2.v, acc[1][1]);
      acc[1][2] = WMMA_BF16(fa1.v, fb3.v, acc[1][2]);
      acc[1][3] = WMMA_BF16(fa3.v, fb3.v, acc[1][3]);
      acc[1][4] = WMMA_BF16(fa1.v, fb4.v, acc[1][4]);
      acc[1][5] = WMMA_BF16(fa3.v, fb4.v, acc[1][5]);
    }
    __syncthreads();   // everyone done with buf[cur] before it is overwritten
  }

  const float* B1 = side ? bh1 : bp1;
  const float* B2 = side ? bh2 : bp2;
  const float* B4 = side ? bh4 : bp4;
  const int n = n0 + l15;
  const float bb1 = B1[n];
  const float bb2 = B2[n];
  const float bb4 = B4[n];

  // C/D layout: VGPR r, lanes 0-15 -> M=r ; lanes 16-31 -> M=8+r ; N = lane&15
  float* outSide = out + (size_t)side * (size_t)MROWS * HDIM;
#pragma unroll
  for (int msub = 0; msub < 2; ++msub) {
#pragma unroll
    for (int r = 0; r < 8; ++r) {
      const int m = m0 + msub * 16 + half * 8 + r;
      float v = (acc[msub][0][r] + bb1) + (acc[msub][1][r] + bb2)
              + (acc[msub][2][r] - acc[msub][3][r])
              + (acc[msub][4][r] + bb4) * (acc[msub][5][r] + bb4);
      outSide[(size_t)m * HDIM + n] = tanhf(v);
    }
  }
}

// ---------------------------------------------------------------------------
extern "C" void kernel_launch(void* const* d_in, const int* in_sizes, int n_in,
                              void* d_out, int out_size, void* d_ws, size_t ws_size,
                              hipStream_t stream) {
  const float* x    = (const float*)d_in[0];
  const int*   pidx = (const int*)d_in[1];
  const int*   hidx = (const int*)d_in[2];

  // Workspace (bf16): [0,32MB) gathered rows (prem_s, prem_e, hypo_s, hypo_e)
  //                   [32,48MB) weights (Wp1..Wp4, Wh1..Wh4)
  unsigned short* ws_a = (unsigned short*)d_ws;
  unsigned short* ws_w = ws_a + (size_t)4 * MROWS * HDIM;

  dim3 gGather(MROWS, 2, 2);
  gather_rows_kernel<<<gGather, 256, 0, stream>>>(x, pidx, hidx, ws_a);

  WPtrs wp;
  wp.w[0] = (const float*)d_in[3];   // Wp1
  wp.w[1] = (const float*)d_in[5];   // Wp2
  wp.w[2] = (const float*)d_in[7];   // Wp3
  wp.w[3] = (const float*)d_in[9];   // Wp4
  wp.w[4] = (const float*)d_in[11];  // Wh1
  wp.w[5] = (const float*)d_in[13];  // Wh2
  wp.w[6] = (const float*)d_in[15];  // Wh3
  wp.w[7] = (const float*)d_in[17];  // Wh4
  dim3 gW((HDIM * HDIM) / (256 * 4), 8, 1);
  convert_w_kernel<<<gW, 256, 0, stream>>>(wp, ws_w);

  dim3 gG(8, MROWS / 32, 2);  // 8 n-groups x 128 m-pairs x 2 sides
  span_gemm_kernel<<<gG, 256, 0, stream>>>(
      ws_a, ws_w,
      (const float*)d_in[4],  (const float*)d_in[6],  (const float*)d_in[10],  // bp1,bp2,bp4
      (const float*)d_in[12], (const float*)d_in[14], (const float*)d_in[18],  // bh1,bh2,bh4
      (float*)d_out);
}
